// GAT2Layer_37082747633989
// MI455X (gfx1250) — compile-verified, matrix operand order checked by reading, not run on hardware
//
#include <hip/hip_runtime.h>
#include <hip/hip_bf16.h>

typedef __attribute__((ext_vector_type(16))) __bf16 v16bf;
typedef __attribute__((ext_vector_type(8)))  float  v8f;

#define IN_DIM 128
#define HEADS 4
#define HIDDEN 32
#define NC 10
#define NEG_SLOPE 0.2f
#define EPS_PYG 1e-16f

// ---------------- utility ----------------

__global__ void fill_kernel(float* __restrict__ p, float v, int n) {
    int i = blockIdx.x * blockDim.x + threadIdx.x;
    if (i < n) p[i] = v;
}

__device__ __forceinline__ void atomicMaxFloat(float* addr, float val) {
    // monotone mapping trick: works for any mix of signs given -FLT_MAX-ish init
    if (val >= 0.0f) atomicMax((int*)addr, __float_as_int(val));
    else             atomicMin((unsigned int*)addr, __float_as_uint(val));
}

__device__ __forceinline__ float leaky(float v) {
    return v > 0.0f ? v : NEG_SLOPE * v;
}

// ---------------- GEMM1: h1 = x @ W1   (M=nN, N=128, K=128, bf16 WMMA) ----------------
// grid.x = nN/16 row tiles; 8 waves/block each own a 16-col tile of the 128 cols.
__global__ __launch_bounds__(256) void gemm1_kernel(const float* __restrict__ X,
                                                    const float* __restrict__ W,
                                                    float* __restrict__ H, int nN) {
    const int wave = threadIdx.x >> 5;
    const int lane = threadIdx.x & 31;
    const int g    = lane >> 4;      // half-wave group
    const int l    = lane & 15;
    const int row0 = blockIdx.x * 16;
    const int col0 = wave * 16;

    v8f c = {0.f,0.f,0.f,0.f,0.f,0.f,0.f,0.f};
    const float* xrow = X + (size_t)(row0 + l) * IN_DIM;

    #pragma unroll
    for (int kt = 0; kt < 4; ++kt) {
        const int kb = kt * 32 + g * 8;
        v16bf a, b;
        #pragma unroll
        for (int e = 0; e < 8; ++e) {
            // A: lane row = row0+l ; elements 0..7 -> K=kb+e, 8..15 -> K=kb+16+e
            a[e]     = (__bf16)xrow[kb + e];
            a[e + 8] = (__bf16)xrow[kb + 16 + e];
            // B: lane col = col0+l ; same K striping, column-major per lane
            b[e]     = (__bf16)W[(size_t)(kb + e) * 128 + col0 + l];
            b[e + 8] = (__bf16)W[(size_t)(kb + 16 + e) * 128 + col0 + l];
        }
        c = __builtin_amdgcn_wmma_f32_16x16x32_bf16(false, a, false, b, (short)0, c, false, false);
    }
    #pragma unroll
    for (int r = 0; r < 8; ++r)
        H[(size_t)(row0 + g * 8 + r) * 128 + col0 + l] = c[r];
}

// ---------------- GEMM2: h2 = out1 @ W2  (N=10 padded to one 16-col WMMA tile) ----------------
__global__ __launch_bounds__(256) void gemm2_kernel(const float* __restrict__ X,
                                                    const float* __restrict__ W,
                                                    float* __restrict__ H, int nN) {
    const int wave = threadIdx.x >> 5;
    const int lane = threadIdx.x & 31;
    const int g    = lane >> 4;
    const int l    = lane & 15;
    const int row0 = (blockIdx.x * 8 + wave) * 16;

    v8f c = {0.f,0.f,0.f,0.f,0.f,0.f,0.f,0.f};
    const int arow = row0 + l;
    const bool aok = (arow < nN);
    const bool bok = (l < NC);

    #pragma unroll
    for (int kt = 0; kt < 4; ++kt) {
        const int kb = kt * 32 + g * 8;
        v16bf a, b;
        #pragma unroll
        for (int e = 0; e < 8; ++e) {
            a[e]     = (__bf16)(aok ? X[(size_t)arow * 128 + kb + e]      : 0.f);
            a[e + 8] = (__bf16)(aok ? X[(size_t)arow * 128 + kb + 16 + e] : 0.f);
            b[e]     = (__bf16)(bok ? W[(size_t)(kb + e) * NC + l]        : 0.f);
            b[e + 8] = (__bf16)(bok ? W[(size_t)(kb + 16 + e) * NC + l]   : 0.f);
        }
        c = __builtin_amdgcn_wmma_f32_16x16x32_bf16(false, a, false, b, (short)0, c, false, false);
    }
    #pragma unroll
    for (int r = 0; r < 8; ++r) {
        const int row = row0 + g * 8 + r;
        if (row < nN && l < NC) H[(size_t)row * NC + l] = c[r];
    }
}

// ---------------- attention coefficients ----------------

__global__ void coef1_kernel(const float* __restrict__ H,
                             const float* __restrict__ asrc, const float* __restrict__ adst,
                             float* __restrict__ as_, float* __restrict__ ad_, int nN) {
    int i = blockIdx.x * blockDim.x + threadIdx.x;
    if (i >= nN) return;
    const float* h = H + (size_t)i * IN_DIM;
    #pragma unroll
    for (int hd = 0; hd < HEADS; ++hd) {
        float s = 0.f, d = 0.f;
        #pragma unroll
        for (int j = 0; j < HIDDEN; ++j) {
            float v = h[hd * HIDDEN + j];
            s += v * asrc[hd * HIDDEN + j];
            d += v * adst[hd * HIDDEN + j];
        }
        as_[i * HEADS + hd] = s;
        ad_[i * HEADS + hd] = d;
    }
}

__global__ void coef2_kernel(const float* __restrict__ H,
                             const float* __restrict__ asrc, const float* __restrict__ adst,
                             float* __restrict__ as_, float* __restrict__ ad_, int nN) {
    int i = blockIdx.x * blockDim.x + threadIdx.x;
    if (i >= nN) return;
    float s = 0.f, d = 0.f;
    #pragma unroll
    for (int j = 0; j < NC; ++j) {
        float v = H[(size_t)i * NC + j];
        s += v * asrc[j];
        d += v * adst[j];
    }
    as_[i] = s;
    ad_[i] = d;
}

// ---------------- edge helpers ----------------

__device__ __forceinline__ void edge_sd(const long long* __restrict__ ei, int e, int nE,
                                        int& s, int& d) {
    if (e < nE) { s = (int)ei[e]; d = (int)ei[nE + e]; }
    else        { s = d = e - nE; }
}

// layer 1: per-head segment max
__global__ void edge_max1_kernel(const long long* __restrict__ ei,
                                 const float* __restrict__ as_, const float* __restrict__ ad_,
                                 float* __restrict__ m, int nE, int nN) {
    int e = blockIdx.x * blockDim.x + threadIdx.x;
    if (e >= nE + nN) return;
    int s, d; edge_sd(ei, e, nE, s, d);
    #pragma unroll
    for (int hd = 0; hd < HEADS; ++hd) {
        float v = leaky(as_[s * HEADS + hd] + ad_[d * HEADS + hd]);
        atomicMaxFloat(&m[d * HEADS + hd], v);
    }
}

__global__ void edge_sum1_kernel(const long long* __restrict__ ei,
                                 const float* __restrict__ as_, const float* __restrict__ ad_,
                                 const float* __restrict__ m, float* __restrict__ z,
                                 int nE, int nN) {
    int e = blockIdx.x * blockDim.x + threadIdx.x;
    if (e >= nE + nN) return;
    int s, d; edge_sd(ei, e, nE, s, d);
    #pragma unroll
    for (int hd = 0; hd < HEADS; ++hd) {
        float v = leaky(as_[s * HEADS + hd] + ad_[d * HEADS + hd]);
        atomicAdd(&z[d * HEADS + hd], __expf(v - m[d * HEADS + hd]));
    }
}

// one wave per (edge, head); lane = feature dim -> coalesced gather + coalesced atomics
__global__ __launch_bounds__(256) void message1_kernel(const long long* __restrict__ ei,
                                                       const float* __restrict__ H,
                                                       const float* __restrict__ as_,
                                                       const float* __restrict__ ad_,
                                                       const float* __restrict__ m,
                                                       const float* __restrict__ z,
                                                       float* __restrict__ out, int nE, int nN) {
    long long tid = (long long)blockIdx.x * blockDim.x + threadIdx.x;
    int lane = threadIdx.x & 31;
    int wid  = (int)(tid >> 5);
    if (wid >= (nE + nN) * HEADS) return;
    int hd = wid & (HEADS - 1);
    int e  = wid >> 2;
    int s, d; edge_sd(ei, e, nE, s, d);
    float v = leaky(as_[s * HEADS + hd] + ad_[d * HEADS + hd]);
    float alpha = __expf(v - m[d * HEADS + hd]) / (z[d * HEADS + hd] + EPS_PYG);
    atomicAdd(&out[(size_t)d * IN_DIM + hd * HIDDEN + lane],
              alpha * H[(size_t)s * IN_DIM + hd * HIDDEN + lane]);
}

// layer 2: single head
__global__ void edge_max2_kernel(const long long* __restrict__ ei,
                                 const float* __restrict__ as_, const float* __restrict__ ad_,
                                 float* __restrict__ m, int nE, int nN) {
    int e = blockIdx.x * blockDim.x + threadIdx.x;
    if (e >= nE + nN) return;
    int s, d; edge_sd(ei, e, nE, s, d);
    atomicMaxFloat(&m[d], leaky(as_[s] + ad_[d]));
}

__global__ void edge_sum2_kernel(const long long* __restrict__ ei,
                                 const float* __restrict__ as_, const float* __restrict__ ad_,
                                 const float* __restrict__ m, float* __restrict__ z,
                                 int nE, int nN) {
    int e = blockIdx.x * blockDim.x + threadIdx.x;
    if (e >= nE + nN) return;
    int s, d; edge_sd(ei, e, nE, s, d);
    float v = leaky(as_[s] + ad_[d]);
    atomicAdd(&z[d], __expf(v - m[d]));
}

// 16 threads per edge, lanes 0..9 carry the 10 classes
__global__ __launch_bounds__(256) void message2_kernel(const long long* __restrict__ ei,
                                                       const float* __restrict__ H,
                                                       const float* __restrict__ as_,
                                                       const float* __restrict__ ad_,
                                                       const float* __restrict__ m,
                                                       const float* __restrict__ z,
                                                       float* __restrict__ out, int nE, int nN) {
    long long tid = (long long)blockIdx.x * blockDim.x + threadIdx.x;
    int c = (int)(tid & 15);
    int e = (int)(tid >> 4);
    if (e >= nE + nN || c >= NC) return;
    int s, d; edge_sd(ei, e, nE, s, d);
    float v = leaky(as_[s] + ad_[d]);
    float alpha = __expf(v - m[d]) / (z[d] + EPS_PYG);
    atomicAdd(&out[(size_t)d * NC + c], alpha * H[(size_t)s * NC + c]);
}

// ---------------- elementwise ----------------

__global__ void elu_bias1_kernel(float* __restrict__ out, const float* __restrict__ b, int total) {
    int i = blockIdx.x * blockDim.x + threadIdx.x;
    if (i >= total) return;
    float v = out[i] + b[i & (IN_DIM - 1)];
    out[i] = v > 0.f ? v : (__expf(v) - 1.f);
}

__global__ void bias2_kernel(float* __restrict__ out, const float* __restrict__ b, int total) {
    int i = blockIdx.x * blockDim.x + threadIdx.x;
    if (i >= total) return;
    out[i] = out[i] + b[i % NC];
}

// ---------------- host launcher ----------------

extern "C" void kernel_launch(void* const* d_in, const int* in_sizes, int n_in,
                              void* d_out, int out_size, void* d_ws, size_t ws_size,
                              hipStream_t stream) {
    const float*     x     = (const float*)d_in[0];
    const long long* ei    = (const long long*)d_in[1];
    const float*     W1    = (const float*)d_in[2];
    const float*     asr1  = (const float*)d_in[3];
    const float*     ads1  = (const float*)d_in[4];
    const float*     b1    = (const float*)d_in[5];
    const float*     W2    = (const float*)d_in[6];
    const float*     asr2  = (const float*)d_in[7];
    const float*     ads2  = (const float*)d_in[8];
    const float*     b2    = (const float*)d_in[9];
    float*           out   = (float*)d_out;

    const int nN = in_sizes[0] / IN_DIM;   // 50000
    const int nE = in_sizes[1] / 2;        // 800000
    const int nT = nE + nN;                // edges incl. self loops

    // workspace layout (floats)
    float* h1   = (float*)d_ws;                       // nN*128
    float* out1 = h1   + (size_t)nN * IN_DIM;         // nN*128
    float* as1  = out1 + (size_t)nN * IN_DIM;         // nN*4
    float* ad1  = as1  + (size_t)nN * HEADS;
    float* m1   = ad1  + (size_t)nN * HEADS;
    float* z1   = m1   + (size_t)nN * HEADS;
    float* h2   = z1   + (size_t)nN * HEADS;          // nN*10
    float* as2  = h2   + (size_t)nN * NC;             // nN
    float* ad2  = as2  + (size_t)nN;
    float* m2   = ad2  + (size_t)nN;
    float* z2   = m2   + (size_t)nN;

    const int B = 256;
    auto blocks = [](long long n, int b) { return (int)((n + b - 1) / b); };

    // ---- layer 1 ----
    fill_kernel<<<blocks((long long)nN * IN_DIM, B), B, 0, stream>>>(out1, 0.f, nN * IN_DIM);
    fill_kernel<<<blocks((long long)nN * HEADS, B), B, 0, stream>>>(m1, -3.4e38f, nN * HEADS);
    fill_kernel<<<blocks((long long)nN * HEADS, B), B, 0, stream>>>(z1, 0.f, nN * HEADS);

    gemm1_kernel<<<nN / 16, 256, 0, stream>>>(x, W1, h1, nN);
    coef1_kernel<<<blocks(nN, B), B, 0, stream>>>(h1, asr1, ads1, as1, ad1, nN);

    edge_max1_kernel<<<blocks(nT, B), B, 0, stream>>>(ei, as1, ad1, m1, nE, nN);
    edge_sum1_kernel<<<blocks(nT, B), B, 0, stream>>>(ei, as1, ad1, m1, z1, nE, nN);
    message1_kernel<<<blocks((long long)nT * HEADS * 32, B), B, 0, stream>>>(
        ei, h1, as1, ad1, m1, z1, out1, nE, nN);

    elu_bias1_kernel<<<blocks((long long)nN * IN_DIM, B), B, 0, stream>>>(out1, b1, nN * IN_DIM);

    // ---- layer 2 ----
    fill_kernel<<<blocks((long long)nN * NC, B), B, 0, stream>>>(out, 0.f, nN * NC);
    fill_kernel<<<blocks(nN, B), B, 0, stream>>>(m2, -3.4e38f, nN);
    fill_kernel<<<blocks(nN, B), B, 0, stream>>>(z2, 0.f, nN);

    gemm2_kernel<<<blocks(nN, 128), 256, 0, stream>>>(out1, W2, h2, nN);
    coef2_kernel<<<blocks(nN, B), B, 0, stream>>>(h2, asr2, ads2, as2, ad2, nN);

    edge_max2_kernel<<<blocks(nT, B), B, 0, stream>>>(ei, as2, ad2, m2, nE, nN);
    edge_sum2_kernel<<<blocks(nT, B), B, 0, stream>>>(ei, as2, ad2, m2, z2, nE, nN);
    message2_kernel<<<blocks((long long)nT * 16, B), B, 0, stream>>>(
        ei, h2, as2, ad2, m2, z2, out, nE, nN);

    bias2_kernel<<<blocks((long long)nN * NC, B), B, 0, stream>>>(out, b2, nN * NC);
}
